// MultiHeadSelfAttention_64029372449080
// MI455X (gfx1250) — matrix operand order repaired and checked by reading, not compiled, not run on hardware
//
#include <hip/hip_runtime.h>

// ===========================================================================
// Fused full-feature self-attention for MI455X (gfx1250, wave32, WMMA).
//
// B=4, S=2048, D=1024 (reference does NOT split heads).
// Pipeline:
//   1) proj_gemm    : q,k,v = in @ W{q,k,v} + b   (fp32 -> f16, WMMA)
//   2) var_finalize : inv_scale = 1/sqrt(var(k, ddof=1))
//   3) qk_gemm      : E[b] = (Q_b @ K_b^T) * inv_scale     (async-LDS staged)
//   4) softmax_rows : P = softmax(E), f16 in place (half-row stride)
//   5) pv_gemm      : X[b] = P_b @ V_b                     (async A staging)
//   6) out_gemm     : out = X @ Wo + bo                    (async A staging)
//
// GEMM shape: block tile 128x128, 8 waves, wave tile 32x64 (2x4 WMMA accs),
// K-step 64 (two 16x16x32 substeps) -> 16 v_wmma per wave per barrier.
// Double-buffered 64KB LDS, one barrier per K-step; pure-copy operands are
// staged with GLOBAL_LOAD_ASYNC_TO_LDS_B128 + s_wait_asynccnt (ASYNCcnt).
//
// Workspace (~112.1 MB):
//   [0,16MB) q f16 (reused as X) | [16,32) k f16 | [32,48) v f16
//   [48,112) E fp32 4x2048x2048 (softmax writes f16 P into each row's first
//            half, row stride 4096 f16) | [112MB,+4KB) 512x{sum,sumsq} | +1 f32
// ===========================================================================

typedef __attribute__((ext_vector_type(16))) _Float16 v16h;
typedef __attribute__((ext_vector_type(8)))  _Float16 v8h;
typedef __attribute__((ext_vector_type(4)))  _Float16 v4h;
typedef __attribute__((ext_vector_type(8)))  float    v8f;

#define WMMA(A, B, C) \
  __builtin_amdgcn_wmma_f32_16x16x32_f16(false, (A), false, (B), (short)0, (C), false, false)

// A fragment (16x32 f16, ISA 7.12.2): lane -> row m=l%16, half h=l/16;
// element j: K = j+8h (j<8), j+8+8h (j>=8)  => two 16B LDS loads + shuffle.
__device__ __forceinline__ v16h fragA(const _Float16* rowbase, int half) {
  v8h lo = *(const v8h*)(rowbase + 8 * half);
  v8h hi = *(const v8h*)(rowbase + 16 + 8 * half);
  return __builtin_shufflevector(lo, hi, 0,1,2,3,4,5,6,7,8,9,10,11,12,13,14,15);
}
// B fragment (32x16 f16): lane -> col n=l%16, half h=l/16; element j: K = j+8h
// (j<8), j+8+8h (j>=8) => two 16B LDS loads, same order as A.
__device__ __forceinline__ v16h fragB(const _Float16* rowbase, int half) {
  v8h lo = *(const v8h*)(rowbase + 8 * half);
  v8h hi = *(const v8h*)(rowbase + 16 + 8 * half);
  return __builtin_shufflevector(lo, hi, 0,1,2,3,4,5,6,7,8,9,10,11,12,13,14,15);
}

// Flat aperture rule (ISA 10.2): low 32 bits of the address are the LDS byte offset.
__device__ __forceinline__ unsigned lds_off(const void* p) {
  return (unsigned)(unsigned long long)(uintptr_t)p;
}
// CDNA5 async DMA: 16B global -> LDS per lane, tracked by ASYNCcnt.
__device__ __forceinline__ void async_b128(unsigned loff, const void* gaddr) {
  asm volatile("global_load_async_to_lds_b128 %0, %1, off"
               :: "v"(loff), "v"(gaddr) : "memory");
}
__device__ __forceinline__ void wait_async0() {
  asm volatile("s_wait_asynccnt 0x0" ::: "memory");
}

// Shared wave-level compute: 32x64 tile, one 64-wide K step (2 substeps).
#define WAVE_COMPUTE(AsBuf, BsBuf)                                            \
  _Pragma("unroll")                                                           \
  for (int s = 0; s < 2; ++s) {                                               \
    v16h a0 = fragA(&(AsBuf)[(wm + mrow) * 64 + 32 * s], half);               \
    v16h a1 = fragA(&(AsBuf)[(wm + 16 + mrow) * 64 + 32 * s], half);          \
    _Pragma("unroll")                                                         \
    for (int j = 0; j < 4; ++j) {                                             \
      v16h bj = fragB(&(BsBuf)[(wn + 16 * j + mrow) * 64 + 32 * s], half);    \
      acc[0][j] = WMMA(a0, bj, acc[0][j]);                                    \
      acc[1][j] = WMMA(a1, bj, acc[1][j]);                                    \
    }                                                                         \
  }

// ---------------------------------------------------------------------------
// 1) proj: Y[M,N]=f16(X@W+bias). fp32 sources -> manual convert staging.
//    grid (M/128, N/128) = (64, 8), block 256. Optional variance partials.
// ---------------------------------------------------------------------------
__global__ __launch_bounds__(256)
void proj_gemm(const float* __restrict__ X, const float* __restrict__ W,
               const float* __restrict__ bias, _Float16* __restrict__ Y,
               int N, int K, float* __restrict__ partials)
{
  __shared__ __align__(64) _Float16 As[2][128 * 64];
  __shared__ __align__(64) _Float16 Bs[2][128 * 64];

  const int bm = blockIdx.x * 128, bn = blockIdx.y * 128;
  const int tid = threadIdx.x, lane = tid & 31, wave = tid >> 5;
  const int wm = (wave & 3) * 32, wn = (wave >> 2) * 64;
  const int mrow = lane & 15, half = lane >> 4;

  v8f acc[2][4] = {};

  auto stage = [&](int nb, int kt) {
    // A: 128x64 fp32 -> f16 (float4 loads, v4h stores)
    for (int c = tid; c < 2048; c += 256) {
      const int r = c >> 4, c4 = (c & 15) * 4;
      float4 f = *(const float4*)&X[(size_t)(bm + r) * K + kt * 64 + c4];
      v4h h = { (_Float16)f.x, (_Float16)f.y, (_Float16)f.z, (_Float16)f.w };
      *(v4h*)&As[nb][r * 64 + c4] = h;
    }
    // B^T: Bs[n][kk] = W[kt*64+kk][bn+n] (float4 along n, scalar transpose)
    const int n4 = (lane) * 4 % 128;        // lane&31 *4
    for (int kk = tid >> 5; kk < 64; kk += 8) {
      float4 f = *(const float4*)&W[(size_t)(kt * 64 + kk) * N + bn + n4];
      Bs[nb][(n4 + 0) * 64 + kk] = (_Float16)f.x;
      Bs[nb][(n4 + 1) * 64 + kk] = (_Float16)f.y;
      Bs[nb][(n4 + 2) * 64 + kk] = (_Float16)f.z;
      Bs[nb][(n4 + 3) * 64 + kk] = (_Float16)f.w;
    }
  };

  const int NIT = K / 64;
  stage(0, 0);
  __syncthreads();
  for (int it = 0; it < NIT; ++it) {
    const int cur = it & 1;
    if (it + 1 < NIT) stage(1 - cur, it + 1);
    WAVE_COMPUTE(As[cur], Bs[cur]);
    __syncthreads();
  }

  float lsum = 0.f, lsq = 0.f;
  #pragma unroll
  for (int i = 0; i < 2; ++i)
    #pragma unroll
    for (int j = 0; j < 4; ++j)
      #pragma unroll
      for (int r = 0; r < 8; ++r) {
        const int m = bm + wm + 16 * i + 8 * half + r;
        const int n = bn + wn + 16 * j + mrow;
        const float v = acc[i][j][r] + bias[n];
        Y[(size_t)m * N + n] = (_Float16)v;
        lsum += v; lsq += v * v;
      }

  if (partials) {  // uniform branch; reuse As LDS for the tree reduction
    float* rs = (float*)(void*)As;
    float* rq = rs + 256;
    rs[tid] = lsum; rq[tid] = lsq;
    __syncthreads();
    for (int s = 128; s > 0; s >>= 1) {
      if (tid < s) { rs[tid] += rs[tid + s]; rq[tid] += rq[tid + s]; }
      __syncthreads();
    }
    if (tid == 0) {
      const int blk = blockIdx.y * gridDim.x + blockIdx.x;
      partials[2 * blk] = rs[0];
      partials[2 * blk + 1] = rq[0];
    }
  }
}

// ---------------------------------------------------------------------------
// 2) inv_scale = 1/sqrt( (sumsq - sum^2/n) / (n-1) )
// ---------------------------------------------------------------------------
__global__ __launch_bounds__(256)
void var_finalize(const float* __restrict__ partials, int nblk, float n_elems,
                  float* __restrict__ inv_scale)
{
  __shared__ float rs[256], rq[256];
  const int t = threadIdx.x;
  float s = 0.f, q = 0.f;
  for (int i = t; i < nblk; i += 256) { s += partials[2 * i]; q += partials[2 * i + 1]; }
  rs[t] = s; rq[t] = q;
  __syncthreads();
  for (int k = 128; k > 0; k >>= 1) {
    if (t < k) { rs[t] += rs[t + k]; rq[t] += rq[t + k]; }
    __syncthreads();
  }
  if (t == 0) {
    const float var = (rq[0] - rs[0] * rs[0] / n_elems) / (n_elems - 1.0f);
    inv_scale[0] = rsqrtf(var);
  }
}

// ---------------------------------------------------------------------------
// 3) E[b] = (Q_b @ K_b^T) * inv_scale. Pure row copies -> both operands use
//    async-to-LDS staging. grid (16,16,4), block 256.
// ---------------------------------------------------------------------------
__global__ __launch_bounds__(256)
void qk_gemm(const _Float16* __restrict__ Q, const _Float16* __restrict__ Kmat,
             const float* __restrict__ inv_scale, float* __restrict__ E)
{
  __shared__ __align__(64) _Float16 As[2][128 * 64];
  __shared__ __align__(64) _Float16 Bs[2][128 * 64];

  const int b = blockIdx.z;
  const _Float16* Qb = Q    + (size_t)b * 2048 * 1024;
  const _Float16* Kb = Kmat + (size_t)b * 2048 * 1024;
  float* Eb = E + (size_t)b * 2048 * 2048;

  const int bm = blockIdx.x * 128, bn = blockIdx.y * 128;
  const int tid = threadIdx.x, lane = tid & 31, wave = tid >> 5;
  const int wm = (wave & 3) * 32, wn = (wave >> 2) * 64;
  const int mrow = lane & 15, half = lane >> 4;

  v8f acc[2][4] = {};

  auto stage = [&](int nb, int kt) {   // 16B chunks: 128 rows x 8 chunks
    for (int c = tid; c < 1024; c += 256) {
      const int r = c >> 3, c8 = (c & 7) * 8;
      async_b128(lds_off(&As[nb][r * 64 + c8]),
                 &Qb[(size_t)(bm + r) * 1024 + kt * 64 + c8]);
      async_b128(lds_off(&Bs[nb][r * 64 + c8]),
                 &Kb[(size_t)(bn + r) * 1024 + kt * 64 + c8]);
    }
  };

  stage(0, 0);
  wait_async0();
  __syncthreads();
  for (int it = 0; it < 16; ++it) {
    const int cur = it & 1;
    if (it + 1 < 16) stage(1 - cur, it + 1);
    WAVE_COMPUTE(As[cur], Bs[cur]);
    wait_async0();
    __syncthreads();
  }

  const float s = inv_scale[0];
  #pragma unroll
  for (int i = 0; i < 2; ++i)
    #pragma unroll
    for (int j = 0; j < 4; ++j)
      #pragma unroll
      for (int r = 0; r < 8; ++r) {
        const int m = bm + wm + 16 * i + 8 * half + r;
        const int n = bn + wn + 16 * j + mrow;
        Eb[(size_t)m * 2048 + n] = acc[i][j][r] * s;
      }
}

// ---------------------------------------------------------------------------
// 4) Row softmax, f16 probabilities written in place (first half of each
//    row's 8KB region; all reads precede the reduction barriers).
// ---------------------------------------------------------------------------
__global__ __launch_bounds__(256)
void softmax_rows(float* __restrict__ E)
{
  float* e = E + (size_t)blockIdx.x * 2048;
  _Float16* p = (_Float16*)(void*)e;
  const int t = threadIdx.x;

  float v[8];
  float m = -3.4e38f;
  #pragma unroll
  for (int i = 0; i < 8; ++i) { v[i] = e[t + 256 * i]; m = fmaxf(m, v[i]); }

  __shared__ float red[256];
  red[t] = m;
  __syncthreads();
  for (int s = 128; s > 0; s >>= 1) {
    if (t < s) red[t] = fmaxf(red[t], red[t + s]);
    __syncthreads();
  }
  m = red[0];
  __syncthreads();

  float sum = 0.f;
  #pragma unroll
  for (int i = 0; i < 8; ++i) { v[i] = __expf(v[i] - m); sum += v[i]; }
  red[t] = sum;
  __syncthreads();
  for (int s = 128; s > 0; s >>= 1) {
    if (t < s) red[t] += red[t + s];
    __syncthreads();
  }
  const float inv = 1.0f / red[0];

  #pragma unroll
  for (int i = 0; i < 8; ++i) p[t + 256 * i] = (_Float16)(v[i] * inv);
}

// ---------------------------------------------------------------------------
// 5) X[b] = P_b @ V_b. P rows (f16, stride 4096) async-staged; V transposed
//    manually. grid (16, 8, 4), block 256. K = 2048 -> 32 iterations.
// ---------------------------------------------------------------------------
__global__ __launch_bounds__(256)
void pv_gemm(const _Float16* __restrict__ P, const _Float16* __restrict__ V,
             _Float16* __restrict__ Xo)
{
  __shared__ __align__(64) _Float16 As[2][128 * 64];
  __shared__ __align__(64) _Float16 Bs[2][128 * 64];

  const int b = blockIdx.z;
  const _Float16* Pb = P  + (size_t)b * 2048 * 4096;
  const _Float16* Vb = V  + (size_t)b * 2048 * 1024;
  _Float16*       Xb = Xo + (size_t)b * 2048 * 1024;

  const int bm = blockIdx.x * 128, bn = blockIdx.y * 128;
  const int tid = threadIdx.x, lane = tid & 31, wave = tid >> 5;
  const int wm = (wave & 3) * 32, wn = (wave >> 2) * 64;
  const int mrow = lane & 15, half = lane >> 4;

  v8f acc[2][4] = {};

  auto stage = [&](int nb, int kt) {
    // A: async row copy, stride 4096
    for (int c = tid; c < 1024; c += 256) {
      const int r = c >> 3, c8 = (c & 7) * 8;
      async_b128(lds_off(&As[nb][r * 64 + c8]),
                 &Pb[(size_t)(bm + r) * 4096 + kt * 64 + c8]);
    }
    // B^T: Bs[n][kk] = V[kt*64+kk][bn+n], v8h loads along n
    const int n8 = (tid & 15) * 8;
    for (int kk = tid >> 4; kk < 64; kk += 16) {
      v8h hv = *(const v8h*)&Vb[(size_t)(kt * 64 + kk) * 1024 + bn + n8];
      #pragma unroll
      for (int j = 0; j < 8; ++j) Bs[nb][(n8 + j) * 64 + kk] = hv[j];
    }
  };

  stage(0, 0);
  wait_async0();
  __syncthreads();
  for (int it = 0; it < 32; ++it) {
    const int cur = it & 1;
    if (it + 1 < 32) stage(1 - cur, it + 1);
    WAVE_COMPUTE(As[cur], Bs[cur]);
    wait_async0();
    __syncthreads();
  }

  #pragma unroll
  for (int i = 0; i < 2; ++i)
    #pragma unroll
    for (int j = 0; j < 4; ++j)
      #pragma unroll
      for (int r = 0; r < 8; ++r) {
        const int m = bm + wm + 16 * i + 8 * half + r;
        const int n = bn + wn + 16 * j + mrow;
        Xb[(size_t)m * 1024 + n] = (_Float16)acc[i][j][r];
      }
}

// ---------------------------------------------------------------------------
// 6) out = fp32( X @ Wo + bo ). X async-staged, Wo transposed+converted.
//    grid (64, 8), block 256.
// ---------------------------------------------------------------------------
__global__ __launch_bounds__(256)
void out_gemm(const _Float16* __restrict__ X, const float* __restrict__ Wo,
              const float* __restrict__ bo, float* __restrict__ Out)
{
  __shared__ __align__(64) _Float16 As[2][128 * 64];
  __shared__ __align__(64) _Float16 Bs[2][128 * 64];

  const int bm = blockIdx.x * 128, bn = blockIdx.y * 128;
  const int tid = threadIdx.x, lane = tid & 31, wave = tid >> 5;
  const int wm = (wave & 3) * 32, wn = (wave >> 2) * 64;
  const int mrow = lane & 15, half = lane >> 4;

  v8f acc[2][4] = {};

  auto stage = [&](int nb, int kt) {
    for (int c = tid; c < 1024; c += 256) {
      const int r = c >> 3, c8 = (c & 7) * 8;
      async_b128(lds_off(&As[nb][r * 64 + c8]),
                 &X[(size_t)(bm + r) * 1024 + kt * 64 + c8]);
    }
    const int n4 = (lane) * 4 % 128;
    for (int kk = tid >> 5; kk < 64; kk += 8) {
      float4 f = *(const float4*)&Wo[(size_t)(kt * 64 + kk) * 1024 + bn + n4];
      Bs[nb][(n4 + 0) * 64 + kk] = (_Float16)f.x;
      Bs[nb][(n4 + 1) * 64 + kk] = (_Float16)f.y;
      Bs[nb][(n4 + 2) * 64 + kk] = (_Float16)f.z;
      Bs[nb][(n4 + 3) * 64 + kk] = (_Float16)f.w;
    }
  };

  stage(0, 0);
  wait_async0();
  __syncthreads();
  for (int it = 0; it < 16; ++it) {
    const int cur = it & 1;
    if (it + 1 < 16) stage(1 - cur, it + 1);
    WAVE_COMPUTE(As[cur], Bs[cur]);
    wait_async0();
    __syncthreads();
  }

  #pragma unroll
  for (int i = 0; i < 2; ++i)
    #pragma unroll
    for (int j = 0; j < 4; ++j)
      #pragma unroll
      for (int r = 0; r < 8; ++r) {
        const int m = bm + wm + 16 * i + 8 * half + r;
        const int n = bn + wn + 16 * j + mrow;
        Out[(size_t)m * 1024 + n] = acc[i][j][r] + bo[n];
      }
}

// ===========================================================================
extern "C" void kernel_launch(void* const* d_in, const int* in_sizes, int n_in,
                              void* d_out, int out_size, void* d_ws, size_t ws_size,
                              hipStream_t stream)
{
  (void)in_sizes; (void)n_in; (void)out_size; (void)ws_size;

  const float* inputs = (const float*)d_in[0];
  const float* Wq = (const float*)d_in[1];
  const float* bq = (const float*)d_in[2];
  const float* Wk = (const float*)d_in[3];
  const float* bk = (const float*)d_in[4];
  const float* Wv = (const float*)d_in[5];
  const float* bv = (const float*)d_in[6];
  const float* Wo = (const float*)d_in[7];
  const float* bo = (const float*)d_in[8];
  float* out = (float*)d_out;

  const size_t MS = (size_t)8192 * 1024;
  char* ws = (char*)d_ws;
  _Float16* q = (_Float16*)ws;                       // 16 MB (reused as X)
  _Float16* k = q + MS;                              // 16 MB
  _Float16* v = k + MS;                              // 16 MB
  float*    E = (float*)(ws + 3 * MS * sizeof(_Float16));          // 64 MB
  float* partials  = (float*)(ws + 3 * MS * sizeof(_Float16)
                                 + (size_t)4 * 2048 * 2048 * sizeof(float));
  float* inv_scale = partials + 2 * 512;

  const dim3 blk(256);
  const dim3 gproj(64, 8);         // 8192/128 x 1024/128
  proj_gemm<<<gproj, blk, 0, stream>>>(inputs, Wq, bq, q, 1024, 1024, nullptr);
  proj_gemm<<<gproj, blk, 0, stream>>>(inputs, Wk, bk, k, 1024, 1024, partials);
  proj_gemm<<<gproj, blk, 0, stream>>>(inputs, Wv, bv, v, 1024, 1024, nullptr);

  var_finalize<<<1, blk, 0, stream>>>(partials, 512, (float)((size_t)4 * 2048 * 1024),
                                      inv_scale);

  const dim3 gqk(16, 16, 4);       // 2048/128 x 2048/128 x B
  qk_gemm<<<gqk, blk, 0, stream>>>(q, k, inv_scale, E);

  softmax_rows<<<4 * 2048, blk, 0, stream>>>(E);

  const dim3 gpv(16, 8, 4);        // 2048/128 x 1024/128 x B
  pv_gemm<<<gpv, blk, 0, stream>>>((const _Float16*)(const void*)E, v, /*X=*/q);

  const dim3 gout(64, 8);
  out_gemm<<<gout, blk, 0, stream>>>(q, Wo, bo, out);
}
